// PN2GeometryEncoder_58463094833335
// MI455X (gfx1250) — compile-verified, matrix operand order checked by reading, not run on hardware
//
#include <hip/hip_runtime.h>
#include <hip/hip_fp16.h>
#include <stdint.h>

typedef __attribute__((ext_vector_type(16))) _Float16 v16h;
typedef __attribute__((ext_vector_type(8)))  float    v8f;
typedef unsigned int u32x4 __attribute__((ext_vector_type(4)));
typedef int          i32x8 __attribute__((ext_vector_type(8)));
typedef int          i32x4 __attribute__((ext_vector_type(4)));

#define BN_EPS 1e-5f
#define FINF   3.402823466e38f

static inline int cdiv(int a, int b) { return (a + b - 1) / b; }
static inline long long cdivll(long long a, long long b) { return (a + b - 1) / b; }

// ----------------------------------------------------------------------------
// Elementwise / conversion kernels
// ----------------------------------------------------------------------------
__global__ void k_f32_to_f16(const float* __restrict__ s, _Float16* __restrict__ d, int n) {
  int i = blockIdx.x * blockDim.x + threadIdx.x;
  if (i < n) d[i] = (_Float16)s[i];
}

// W [K,N] f32 row-major -> Wh [Kpad,N] f16, zero rows for k >= K
__global__ void k_convert_weight(const float* __restrict__ W, int K, int N,
                                 _Float16* __restrict__ Wh, int Kpad) {
  int i = blockIdx.x * blockDim.x + threadIdx.x;
  if (i >= Kpad * N) return;
  int k = i / N, n = i % N;
  Wh[i] = (k < K) ? (_Float16)W[k * N + n] : (_Float16)0.0f;
}

__global__ void k_copy_f32(const float* __restrict__ s, float* __restrict__ d, int n) {
  int i = blockIdx.x * blockDim.x + threadIdx.x;
  if (i < n) d[i] = s[i];
}

__global__ void k_copy_h(const _Float16* __restrict__ src, int lds_, int rows, int cols,
                         _Float16* __restrict__ dst, int ldd, int coff) {
  int gid = blockIdx.x * blockDim.x + threadIdx.x;
  if (gid >= rows * cols) return;
  int r = gid / cols, c = gid % cols;
  dst[(size_t)r * ldd + coff + c] = src[(size_t)r * lds_ + c];
}

__global__ void k_gather3(const float* __restrict__ pos, const int* __restrict__ idx,
                          int M, float* __restrict__ out) {
  int gid = blockIdx.x * blockDim.x + threadIdx.x;
  if (gid >= M * 3) return;
  int i = gid / 3, c = gid % 3;
  out[gid] = pos[idx[i] * 3 + c];
}

// ----------------------------------------------------------------------------
// WMMA GEMM, fused epilogue.
//   A : [M, lda] f16 row-major (K zero-padded, K % 32 == 0)
//   W : [K, ldw] f16 row-major
//   mode 0: y = relu(h * gamma*rsqrt(1+eps) + beta)   (eval BN)
//   mode 1: y = h + beta (bias)
// 256 threads = 8 waves as 4(M) x 2(N); wave tile 16x32 (2 accumulators, two
// v_wmma_f32_16x16x32_f16 per K-step). W tile (32K x 64N) is DMA'd to LDS by
// the Tensor Data Mover (wave 0: tensor_load_to_lds + s_wait_tensorcnt), then
// transposed in LDS so B-frag reads merge into ds_load_b128.
// ----------------------------------------------------------------------------
__global__ void __launch_bounds__(256) k_wmma_gemm(
    const _Float16* __restrict__ A, int lda,
    const _Float16* __restrict__ W, int ldw,
    int M, int N, int K,
    const float* __restrict__ gamma, const float* __restrict__ beta,
    int mode,
    float* __restrict__ Cf, int ldc,
    _Float16* __restrict__ Ch, int ldh)
{
  __shared__ _Float16 Wr[32][64];   // raw k-major W tile (TDM destination)
  __shared__ _Float16 Bs[64][32];   // transposed [n][k] tile for frag reads

  const int lane  = threadIdx.x & 31;
  const int wave  = threadIdx.x >> 5;
  const int l16   = lane & 15;
  const int lh    = lane >> 4;
  const int wm    = wave >> 1;       // 0..3 along M
  const int wn    = wave & 1;        // 0..1 along N
  const int nblk  = blockIdx.x * 64; // block N base
  const int nw    = wn * 32;         // wave N offset in block
  const int mt0   = blockIdx.y * 64 + wm * 16;
  const bool active = (mt0 < M);
  const int mtile = active ? mt0 : 0;   // inactive waves recompute row 0 (no store)

  v8f acc0 = {}, acc1 = {};
  const _Float16* Arow = A + (size_t)(mtile + l16) * lda;

  for (int kk = 0; kk < K; kk += 32) {
#if __has_builtin(__builtin_amdgcn_tensor_load_to_lds) && \
    __has_builtin(__builtin_amdgcn_s_wait_tensorcnt)
    if (wave == 0) {
      // Build 2D tensor-DMA descriptor (D#): 32 rows x 64 halves, row stride ldw
      unsigned long long ga =
          (unsigned long long)(uintptr_t)(W + (size_t)kk * ldw + nblk);
      unsigned int ldsoff = (unsigned int)(uintptr_t)(&Wr[0][0]); // LDS aperture low32
      u32x4 g0;
      g0[0] = 1u;                                   // count=1, user mode
      g0[1] = ldsoff;                               // lds_addr
      g0[2] = (unsigned int)ga;                     // global_addr[31:0]
      g0[3] = (unsigned int)((ga >> 32) & 0x01FFFFFFu) | (2u << 30); // [56:32] | type=2
      i32x8 g1;
      g1[0] = 0x00010000;                           // wg_mask=0, data_size=1 (2B)
      g1[1] = (int)(((unsigned)ldw & 0xFFFFu) << 16);       // tensor_dim0 lo16
      g1[2] = (int)(((unsigned)ldw >> 16) | (32u << 16));   // dim0 hi16 | tensor_dim1 lo16=32
      g1[3] = (int)(64u << 16);                     // dim1 hi16=0 | tile_dim0=64
      g1[4] = 32;                                   // tile_dim1=32, tile_dim2=0
      g1[5] = ldw;                                  // tensor_dim0_stride lo32
      g1[6] = 0;                                    // stride0 hi16 | stride1 lo16
      g1[7] = 0;                                    // stride1 hi32
      i32x4 z4 = {0, 0, 0, 0};
      i32x8 z8 = {0, 0, 0, 0, 0, 0, 0, 0};
      __builtin_amdgcn_tensor_load_to_lds(g0, g1, z4, z4, z8, 0);
      __builtin_amdgcn_s_wait_tensorcnt(0);
    }
    __syncthreads();
#else
    for (int e = threadIdx.x; e < 2048; e += 256) {
      int n = e & 63, k = e >> 6;
      Wr[k][n] = W[(size_t)(kk + k) * ldw + nblk + n];
    }
    __syncthreads();
#endif
    // transpose raw tile -> [n][k] so per-lane frag reads are contiguous pairs
    for (int e = threadIdx.x; e < 2048; e += 256) {
      int n = e & 63, k = e >> 6;
      Bs[n][k] = Wr[k][n];
    }
    __syncthreads();
    if (kk + 32 < K)
      __builtin_prefetch(W + (size_t)(kk + 32) * ldw + nblk, 0, 1); // global_prefetch_b8

    // ISA A-frag layout (16-bit A 16x32): lane m=l%16, khalf=l/16;
    // vgpr v holds K pair at k0 = (v/4)*16 + khalf*8 + (v%4)*2.
    v16h a, b0, b1;
#pragma unroll
    for (int v = 0; v < 8; ++v) {
      int k0 = ((v >> 2) << 4) + (lh << 3) + ((v & 3) << 1);
      a[2 * v]      = Arow[kk + k0];
      a[2 * v + 1]  = Arow[kk + k0 + 1];
      b0[2 * v]     = Bs[nw + l16][k0];
      b0[2 * v + 1] = Bs[nw + l16][k0 + 1];
      b1[2 * v]     = Bs[nw + 16 + l16][k0];
      b1[2 * v + 1] = Bs[nw + 16 + l16][k0 + 1];
    }
    acc0 = __builtin_amdgcn_wmma_f32_16x16x32_f16(false, a, false, b0,
                                                  (short)0, acc0, false, false);
    acc1 = __builtin_amdgcn_wmma_f32_16x16x32_f16(false, a, false, b1,
                                                  (short)0, acc1, false, false);
    __syncthreads();
  }

  if (active) {
#pragma unroll
    for (int t = 0; t < 2; ++t) {
      const int n = nblk + nw + t * 16 + l16;
      float sc = 1.0f, sh = 0.0f;
      if (mode == 0) { sc = gamma[n] * rsqrtf(1.0f + BN_EPS); sh = beta[n]; }
      else           { sh = beta[n]; }
      const v8f acc = t ? acc1 : acc0;
#pragma unroll
      for (int r = 0; r < 8; ++r) {
        int m   = mtile + (lh << 3) + r;     // C layout: m = 8*(lane/16)+vgpr
        float h = acc[r];
        float y = (mode == 0) ? fmaxf(h * sc + sh, 0.0f) : (h + sh);
        if (Cf) Cf[(size_t)m * ldc + n] = y;
        if (Ch) Ch[(size_t)m * ldh + n] = (_Float16)y;
      }
    }
  }
}

// ----------------------------------------------------------------------------
// Farthest point sampling. One block (1024 threads), sequential scan with a
// block-wide argmax reduction per step. idx[s] = carry's `last` (start 0).
// ----------------------------------------------------------------------------
__global__ void __launch_bounds__(1024) k_fps(const float* __restrict__ pos,
                                              int N, int S, int* __restrict__ idx)
{
  __shared__ float sv[1024];
  __shared__ int   si[1024];
  __shared__ int   lastIdx;
  const int t = threadIdx.x;
  float md[16];
#pragma unroll
  for (int i = 0; i < 16; ++i) md[i] = FINF;
  if (t == 0) lastIdx = 0;
  __syncthreads();

  for (int s = 0; s < S; ++s) {
    const int li = lastIdx;
    const float lx = pos[li * 3], ly = pos[li * 3 + 1], lz = pos[li * 3 + 2];
    float bv = -1.0f; int bi = 0;
#pragma unroll
    for (int i = 0; i < 16; ++i) {
      int p = t + (i << 10);
      if (p < N) {
        float dx = pos[p * 3] - lx, dy = pos[p * 3 + 1] - ly, dz = pos[p * 3 + 2] - lz;
        float d = dx * dx + dy * dy + dz * dz;
        if (d < md[i]) md[i] = d;
        if (md[i] > bv) { bv = md[i]; bi = p; }
      }
    }
    sv[t] = bv; si[t] = bi;
    __syncthreads();
    for (int off = 512; off > 0; off >>= 1) {
      if (t < off && sv[t + off] > sv[t]) { sv[t] = sv[t + off]; si[t] = si[t + off]; }
      __syncthreads();
    }
    if (t == 0) { idx[s] = li; lastIdx = si[0]; }
    __syncthreads();
  }
}

// ----------------------------------------------------------------------------
// Ball query: k nearest within radius. One thread per center (order-invariant
// downstream: max-pool / interp).
// ----------------------------------------------------------------------------
__global__ void k_ball_query(const float* __restrict__ pos, int Np,
                             const float* __restrict__ ctr, int M,
                             float r2, int K,
                             int* __restrict__ nbr, int* __restrict__ nv)
{
  int i = blockIdx.x * blockDim.x + threadIdx.x;
  if (i >= M) return;
  const float cx = ctr[i * 3], cy = ctr[i * 3 + 1], cz = ctr[i * 3 + 2];
  float bd[64]; int bi[64];
  int cnt = 0, worst = 0; float worstd = -1.0f;
  for (int p = 0; p < Np; ++p) {
    float dx = pos[p * 3] - cx, dy = pos[p * 3 + 1] - cy, dz = pos[p * 3 + 2] - cz;
    float d = dx * dx + dy * dy + dz * dz;
    if (d <= r2) {
      if (cnt < K) {
        bd[cnt] = d; bi[cnt] = p;
        if (d > worstd) { worstd = d; worst = cnt; }
        ++cnt;
      } else if (d < worstd) {
        bd[worst] = d; bi[worst] = p;
        worstd = -1.0f;
        for (int q = 0; q < K; ++q)
          if (bd[q] > worstd) { worstd = bd[q]; worst = q; }
      }
    }
  }
  nv[i] = cnt;
  for (int q = 0; q < K; ++q) nbr[i * K + q] = (q < cnt) ? bi[q] : 0;
}

// Build padded f16 GEMM rows: [x_j (C) | pos_j - ctr_i (3) | zeros]
__global__ void k_build_rows(const _Float16* __restrict__ X, int C, int ldx,
                             const float* __restrict__ pos,
                             const float* __restrict__ ctr,
                             const int* __restrict__ nbr, const int* __restrict__ nv,
                             int M, int K, _Float16* __restrict__ H, int ldr)
{
  long long gid = (long long)blockIdx.x * blockDim.x + threadIdx.x;
  long long total = (long long)M * K * ldr;
  if (gid >= total) return;
  int c = (int)(gid % ldr);
  long long row = gid / ldr;
  int i = (int)(row / K), s = (int)(row % K);
  _Float16 v = (_Float16)0.0f;
  if (s < nv[i]) {
    int j = nbr[i * K + s];
    if (c < C) v = X[(size_t)j * ldx + c];
    else if (c < C + 3) {
      int d = c - C;
      v = (_Float16)(pos[j * 3 + d] - ctr[i * 3 + d]);
    }
  }
  H[gid] = v;
}

// Masked max over K neighbor slots (nv[i] >= 1: center is in its own ball)
__global__ void k_maxpool_slots(const _Float16* __restrict__ H, int ldh,
                                const int* __restrict__ nv,
                                int M, int K, int N,
                                _Float16* __restrict__ out, int ldo)
{
  int gid = blockIdx.x * blockDim.x + threadIdx.x;
  if (gid >= M * N) return;
  int i = gid / N, n = gid % N;
  int cnt = nv[i];
  float best = -FINF;
  for (int s = 0; s < cnt; ++s)
    best = fmaxf(best, (float)H[(size_t)(i * K + s) * ldh + n]);
  out[(size_t)i * ldo + n] = (_Float16)best;
}

// Global max over M rows -> padded [16, N] (row 0 = max, rows 1..15 zero)
__global__ void k_rowmax_pad16(const _Float16* __restrict__ X, int ldx, int M, int N,
                               _Float16* __restrict__ out)
{
  int gid = blockIdx.x * blockDim.x + threadIdx.x;
  if (gid >= 16 * N) return;
  int r = gid / N, n = gid % N;
  float v = 0.0f;
  if (r == 0) {
    float best = -FINF;
    for (int m = 0; m < M; ++m) best = fmaxf(best, (float)X[(size_t)m * ldx + n]);
    v = best;
  }
  out[gid] = (_Float16)v;
}

// Top-3 nearest sources for each destination point + inverse-d2 weights
__global__ void k_knn3(const float* __restrict__ ps, int Ns,
                       const float* __restrict__ pd, int Nd,
                       int* __restrict__ idx3, float* __restrict__ w3)
{
  int i = blockIdx.x * blockDim.x + threadIdx.x;
  if (i >= Nd) return;
  const float x = pd[i * 3], y = pd[i * 3 + 1], z = pd[i * 3 + 2];
  float d0 = FINF, d1 = FINF, d2 = FINF;
  int   i0 = 0,   i1 = 0,    i2 = 0;
  for (int p = 0; p < Ns; ++p) {
    float dx = ps[p * 3] - x, dy = ps[p * 3 + 1] - y, dz = ps[p * 3 + 2] - z;
    float d = dx * dx + dy * dy + dz * dz;
    if (d < d0)      { d2 = d1; i2 = i1; d1 = d0; i1 = i0; d0 = d; i0 = p; }
    else if (d < d1) { d2 = d1; i2 = i1; d1 = d;  i1 = p; }
    else if (d < d2) { d2 = d;  i2 = p; }
  }
  idx3[i * 3] = i0; idx3[i * 3 + 1] = i1; idx3[i * 3 + 2] = i2;
  w3[i * 3]     = 1.0f / fmaxf(d0, 1e-16f);
  w3[i * 3 + 1] = 1.0f / fmaxf(d1, 1e-16f);
  w3[i * 3 + 2] = 1.0f / fmaxf(d2, 1e-16f);
}

// Inverse-distance weighted 3-NN interpolation, writes into concat buffer
__global__ void k_interp(const _Float16* __restrict__ X, int ldx,
                         const int* __restrict__ idx3, const float* __restrict__ w3,
                         int Nd, int C,
                         _Float16* __restrict__ out, int ldo, int coff)
{
  long long gid = (long long)blockIdx.x * blockDim.x + threadIdx.x;
  if (gid >= (long long)Nd * C) return;
  int i = (int)(gid / C), c = (int)(gid % C);
  float w0 = w3[i * 3], w1 = w3[i * 3 + 1], w2 = w3[i * 3 + 2];
  float v = w0 * (float)X[(size_t)idx3[i * 3]     * ldx + c]
          + w1 * (float)X[(size_t)idx3[i * 3 + 1] * ldx + c]
          + w2 * (float)X[(size_t)idx3[i * 3 + 2] * ldx + c];
  out[(size_t)i * ldo + coff + c] = (_Float16)(v / (w0 + w1 + w2));
}

// Tail of fp0 concat rows: cols 256..258 = pos (f16), 259..287 = zero padding
__global__ void k_cat0_tail(const float* __restrict__ pos, int Nd,
                            _Float16* __restrict__ cat, int ldr)
{
  int gid = blockIdx.x * blockDim.x + threadIdx.x;
  if (gid >= Nd * 32) return;
  int i = gid / 32, c2 = gid % 32;
  cat[(size_t)i * ldr + 256 + c2] =
      (c2 < 3) ? (_Float16)pos[i * 3 + c2] : (_Float16)0.0f;
}

// ----------------------------------------------------------------------------
// Host driver
// ----------------------------------------------------------------------------
extern "C" void kernel_launch(void* const* d_in, const int* in_sizes, int n_in,
                              void* d_out, int out_size, void* d_ws, size_t ws_size,
                              hipStream_t stream)
{
  (void)in_sizes; (void)n_in; (void)out_size; (void)ws_size;
  const int NPTS = 16384, NB = 4, N1 = 512, N2 = 128, K1 = 32, K2 = 64;
  const float* pts = (const float*)d_in[0];
  auto F = [&](int i) { return (const float*)d_in[i]; };

  // bump allocator over workspace
  char* base = (char*)d_ws; size_t off = 0;
  auto alloc = [&](size_t bytes) -> void* {
    void* p = base + off; off += (bytes + 255) & ~(size_t)255; return p;
  };

  // ---- convert weights to f16 with K zero-padded to multiples of 32 ----
  struct WT { int idx, K, N, Kpad; };
  const WT wts[14] = {
    {1, 6, 64, 32}, {4, 64, 64, 64}, {7, 64, 128, 64}, {10, 128, 256, 128},
    {13, 259, 128, 288}, {16, 128, 128, 128}, {19, 128, 256, 128}, {22, 256, 256, 256},
    {25, 256, 512, 256}, {28, 512, 256, 512},
    {30, 512, 256, 512}, {33, 256, 256, 256},
    {36, 259, 256, 288}, {39, 256, 256, 256}
  };
  _Float16* wh[14];
  for (int i = 0; i < 14; ++i) {
    wh[i] = (_Float16*)alloc((size_t)wts[i].Kpad * wts[i].N * 2);
    int tot = wts[i].Kpad * wts[i].N;
    k_convert_weight<<<cdiv(tot, 256), 256, 0, stream>>>(F(wts[i].idx), wts[i].K,
                                                         wts[i].N, wh[i], wts[i].Kpad);
  }

  // ---- per-batch scratch (reused; batches run sequentially on stream) ----
  _Float16* pos_h = (_Float16*)alloc((size_t)NPTS * 3 * 2);
  int*   idx1 = (int*)alloc(N1 * 4);
  float* pos1 = (float*)alloc(N1 * 3 * 4);
  int*   nbr1 = (int*)alloc((size_t)N1 * K1 * 4);
  int*   nv1  = (int*)alloc(N1 * 4);
  _Float16* rows1 = (_Float16*)alloc((size_t)N1 * K1 * 32 * 2);
  _Float16* h1a = (_Float16*)alloc((size_t)16384 * 64 * 2);
  _Float16* h1b = (_Float16*)alloc((size_t)16384 * 64 * 2);
  _Float16* h1c = (_Float16*)alloc((size_t)16384 * 128 * 2);
  _Float16* x1pre = (_Float16*)alloc((size_t)N1 * 128 * 2);
  _Float16* x1    = (_Float16*)alloc((size_t)N1 * 256 * 2);
  int*   idx2 = (int*)alloc(N2 * 4);
  float* pos2 = (float*)alloc(N2 * 3 * 4);
  int*   nbr2 = (int*)alloc((size_t)N2 * K2 * 4);
  int*   nv2  = (int*)alloc(N2 * 4);
  _Float16* rows2 = (_Float16*)alloc((size_t)N2 * K2 * 288 * 2);
  _Float16* h2a = (_Float16*)alloc((size_t)8192 * 128 * 2);
  _Float16* h2b = (_Float16*)alloc((size_t)8192 * 128 * 2);
  _Float16* h2c = (_Float16*)alloc((size_t)8192 * 256 * 2);
  _Float16* x2pre = (_Float16*)alloc((size_t)N2 * 256 * 2);
  _Float16* x2    = (_Float16*)alloc((size_t)N2 * 256 * 2);
  _Float16* gin   = (_Float16*)alloc((size_t)16 * 256 * 2);
  _Float16* gh1   = (_Float16*)alloc((size_t)16 * 512 * 2);
  float*    gout  = (float*)alloc((size_t)16 * 256 * 4);
  int*   idx3a = (int*)alloc((size_t)N1 * 3 * 4);
  float* w3a   = (float*)alloc((size_t)N1 * 3 * 4);
  _Float16* cat1 = (_Float16*)alloc((size_t)N1 * 512 * 2);
  _Float16* f1a  = (_Float16*)alloc((size_t)N1 * 256 * 2);
  _Float16* x1fp = (_Float16*)alloc((size_t)N1 * 256 * 2);
  int*   idx3b = (int*)alloc((size_t)NPTS * 3 * 4);
  float* w3b   = (float*)alloc((size_t)NPTS * 3 * 4);
  _Float16* cat0 = (_Float16*)alloc((size_t)NPTS * 288 * 2);
  _Float16* f0a  = (_Float16*)alloc((size_t)NPTS * 256 * 2);

  auto gemm = [&](const _Float16* A, int lda, const _Float16* W, int ldw,
                  int M, int Nn, int K, const float* g, const float* b, int mode,
                  float* Cf, int ldc, _Float16* Ch, int ldh) {
    dim3 grid(Nn / 64, cdiv(M, 64));
    k_wmma_gemm<<<grid, 256, 0, stream>>>(A, lda, W, ldw, M, Nn, K, g, b, mode,
                                          Cf, ldc, Ch, ldh);
  };

  float* outF = (float*)d_out;
  float* outG = outF + (size_t)NB * NPTS * 256;

  for (int b = 0; b < NB; ++b) {
    const float* pos = pts + (size_t)b * NPTS * 3;
    k_f32_to_f16<<<cdiv(NPTS * 3, 256), 256, 0, stream>>>(pos, pos_h, NPTS * 3);

    // ---------------- SA1 ----------------
    k_fps<<<1, 1024, 0, stream>>>(pos, NPTS, N1, idx1);
    k_gather3<<<cdiv(N1 * 3, 256), 256, 0, stream>>>(pos, idx1, N1, pos1);
    k_ball_query<<<cdiv(N1, 128), 128, 0, stream>>>(pos, NPTS, pos1, N1,
                                                    0.04f, K1, nbr1, nv1);
    {
      long long tot = (long long)N1 * K1 * 32;
      k_build_rows<<<(unsigned)cdivll(tot, 256), 256, 0, stream>>>(
          pos_h, 3, 3, pos, pos1, nbr1, nv1, N1, K1, rows1, 32);
    }
    gemm(rows1, 32, wh[0], 64, N1 * K1, 64, 32, F(2), F(3), 0, nullptr, 0, h1a, 64);
    gemm(h1a, 64, wh[1], 64, 16384, 64, 64, F(5), F(6), 0, nullptr, 0, h1b, 64);
    gemm(h1b, 64, wh[2], 128, 16384, 128, 64, F(8), F(9), 0, nullptr, 0, h1c, 128);
    k_maxpool_slots<<<cdiv(N1 * 128, 256), 256, 0, stream>>>(h1c, 128, nv1, N1, K1,
                                                             128, x1pre, 128);
    gemm(x1pre, 128, wh[3], 256, N1, 256, 128, F(11), F(12), 0, nullptr, 0, x1, 256);

    // ---------------- SA2 ----------------
    k_fps<<<1, 1024, 0, stream>>>(pos1, N1, N2, idx2);
    k_gather3<<<cdiv(N2 * 3, 256), 256, 0, stream>>>(pos1, idx2, N2, pos2);
    k_ball_query<<<1, 128, 0, stream>>>(pos1, N1, pos2, N2, 0.16f, K2, nbr2, nv2);
    {
      long long tot = (long long)N2 * K2 * 288;
      k_build_rows<<<(unsigned)cdivll(tot, 256), 256, 0, stream>>>(
          x1, 256, 256, pos1, pos2, nbr2, nv2, N2, K2, rows2, 288);
    }
    gemm(rows2, 288, wh[4], 128, N2 * K2, 128, 288, F(14), F(15), 0, nullptr, 0, h2a, 128);
    gemm(h2a, 128, wh[5], 128, 8192, 128, 128, F(17), F(18), 0, nullptr, 0, h2b, 128);
    gemm(h2b, 128, wh[6], 256, 8192, 256, 128, F(20), F(21), 0, nullptr, 0, h2c, 256);
    k_maxpool_slots<<<cdiv(N2 * 256, 256), 256, 0, stream>>>(h2c, 256, nv2, N2, K2,
                                                             256, x2pre, 256);
    gemm(x2pre, 256, wh[7], 256, N2, 256, 256, F(23), F(24), 0, nullptr, 0, x2, 256);

    // ---------------- global vector ----------------
    k_rowmax_pad16<<<cdiv(16 * 256, 256), 256, 0, stream>>>(x2, 256, N2, 256, gin);
    gemm(gin, 256, wh[8], 512, 16, 512, 256, F(26), F(27), 0, nullptr, 0, gh1, 512);
    gemm(gh1, 512, wh[9], 256, 16, 256, 512, nullptr, F(29), 1, gout, 256, nullptr, 0);
    k_copy_f32<<<1, 256, 0, stream>>>(gout, outG + (size_t)b * 256, 256);

    // ---------------- FP1: x2 (pos2) -> pos1 ----------------
    k_knn3<<<cdiv(N1, 128), 128, 0, stream>>>(pos2, N2, pos1, N1, idx3a, w3a);
    k_interp<<<cdiv(N1 * 256, 256), 256, 0, stream>>>(x2, 256, idx3a, w3a, N1, 256,
                                                      cat1, 512, 0);
    k_copy_h<<<cdiv(N1 * 256, 256), 256, 0, stream>>>(x1, 256, N1, 256, cat1, 512, 256);
    gemm(cat1, 512, wh[10], 256, N1, 256, 512, F(31), F(32), 0, nullptr, 0, f1a, 256);
    gemm(f1a, 256, wh[11], 256, N1, 256, 256, F(34), F(35), 0, nullptr, 0, x1fp, 256);

    // ---------------- FP0: x1fp (pos1) -> pos ----------------
    k_knn3<<<cdiv(NPTS, 256), 256, 0, stream>>>(pos1, N1, pos, NPTS, idx3b, w3b);
    {
      long long tot = (long long)NPTS * 256;
      k_interp<<<(unsigned)cdivll(tot, 256), 256, 0, stream>>>(
          x1fp, 256, idx3b, w3b, NPTS, 256, cat0, 288, 0);
    }
    k_cat0_tail<<<cdiv(NPTS * 32, 256), 256, 0, stream>>>(pos, NPTS, cat0, 288);
    gemm(cat0, 288, wh[12], 256, NPTS, 256, 288, F(37), F(38), 0, nullptr, 0, f0a, 256);
    gemm(f0a, 256, wh[13], 256, NPTS, 256, 256, nullptr, F(40), 1,
         outF + (size_t)b * NPTS * 256, 256, nullptr, 0);
  }
}
